// PSP_82162724372866
// MI455X (gfx1250) — compile-verified, matrix operand order checked by reading, not run on hardware
//
#include <hip/hip_runtime.h>
#include <hip/hip_bf16.h>

// ---------------- CDNA5 (gfx1250) PSP kernel pipeline -----------------------
// bf16 WMMA (v_wmma_f32_16x16x32_bf16) for all GEMMs, f32 accumulate,
// f32 for reductions / normalize / threshold / LayerNorm.
// GEMM waves compute 32x32 macro-tiles (2x2 register blocking): 4 independent
// accumulators break the WMMA->WMMA RAW chain and double FLOP/byte.

#define B_ 8
#define T_ 2048
#define D_ 256
#define EPS_NORM 1e-8f
#define EPS_LN   1e-6f

typedef __bf16 bf16_t;
typedef __attribute__((ext_vector_type(16))) __bf16 v16bf;
typedef __attribute__((ext_vector_type(8)))  __bf16 v8bf;
typedef __attribute__((ext_vector_type(8)))  float  v8f;

// Fragment loader for A (and K-contiguous "transposed B") operands.
// Lane L = (h<<4)|r ; element e -> k = ((e>>3)<<4) + h*8 + (e&7).
// The 16 bf16 elements are two contiguous 8-element (16-byte) runs:
// rows are 16B-aligned, so these lower to global_load_b128.
__device__ __forceinline__ v16bf load_frag(const bf16_t* base, int ld) {
  int lane = threadIdx.x & 31;
  int r = lane & 15, h = lane >> 4;
  const bf16_t* p = base + (size_t)r * ld + h * 8;
  v8bf lo = *(const v8bf*)(p);
  v8bf hi = *(const v8bf*)(p + 16);
  v16bf f;
#pragma unroll
  for (int i = 0; i < 8; ++i) { f[i] = lo[i]; f[i + 8] = hi[i]; }
  return f;
}

__device__ __forceinline__ v8f wmma_bf16(v16bf a, v16bf b, v8f c) {
  return __builtin_amdgcn_wmma_f32_16x16x32_bf16(
      /*neg_a=*/false, a, /*neg_b=*/false, b,
      /*c_mod=*/(short)0, c, /*reuse_a=*/false, /*reuse_b=*/false);
}

// 32x32 macro-tile GEMM core: C[32x32] += A[32xK] * Bt[32xK]^T
// (Bt is the N-major, K-contiguous layout). 4 independent accumulators.
struct Acc4 { v8f c00, c01, c10, c11; };

__device__ __forceinline__ Acc4 gemm_32x32(const bf16_t* __restrict__ A,
                                           const bf16_t* __restrict__ Bt,
                                           int ld, int K) {
  Acc4 t;
  t.c00 = {}; t.c01 = {}; t.c10 = {}; t.c11 = {};
  for (int k0 = 0; k0 < K; k0 += 32) {
    v16bf a0 = load_frag(A + k0, ld);
    v16bf a1 = load_frag(A + (size_t)16 * ld + k0, ld);
    v16bf b0 = load_frag(Bt + k0, ld);
    v16bf b1 = load_frag(Bt + (size_t)16 * ld + k0, ld);
    t.c00 = wmma_bf16(a0, b0, t.c00);
    t.c01 = wmma_bf16(a0, b1, t.c01);
    t.c10 = wmma_bf16(a1, b0, t.c10);
    t.c11 = wmma_bf16(a1, b1, t.c11);
  }
  return t;
}

// C/D layout: VGPR r, lane L=(h<<4)|nl -> element (M = r + 8*h, N = nl).
__device__ __forceinline__ void tile_coords(int& nl, int& hl) {
  int lane = threadIdx.x & 31;
  nl = lane & 15; hl = lane >> 4;
}

// ---- 1) fp32 -> bf16 copies of activations; weights transposed to Wt[h][d].
__global__ void k_convert(const float* __restrict__ a_fea, const float* __restrict__ v_fea,
                          const float* __restrict__ Wv1, const float* __restrict__ Wv2,
                          const float* __restrict__ Wa1, const float* __restrict__ Wa2,
                          bf16_t* __restrict__ afea_bf, bf16_t* __restrict__ vfea_bf,
                          bf16_t* __restrict__ Wv1T, bf16_t* __restrict__ Wv2T,
                          bf16_t* __restrict__ Wa1T, bf16_t* __restrict__ Wa2T) {
  int i = blockIdx.x * 256 + threadIdx.x;
  int n_act = B_ * T_ * D_;
  if (i < n_act) {
    afea_bf[i] = (bf16_t)a_fea[i];
    vfea_bf[i] = (bf16_t)v_fea[i];
  }
  if (i < D_ * D_) {
    int d = i / D_, h = i % D_;          // W[d][h] -> Wt[h][d]
    size_t o = (size_t)h * D_ + d;
    Wv1T[o] = (bf16_t)Wv1[i];
    Wv2T[o] = (bf16_t)Wv2[i];
    Wa1T[o] = (bf16_t)Wa1[i];
    Wa2T[o] = (bf16_t)Wa2[i];
  }
}

// ---- 2) four projections relu(X @ W), one 32x32 tile per wave, K=256.
// which: 0 -> v_b1 (stored transposed [b][h][t]), 1 -> v_b2 (row-major),
//        2 -> a_b1 (row-major),                   3 -> a_b2 (transposed).
__global__ __launch_bounds__(32)
void k_proj(const bf16_t* __restrict__ vfea, const bf16_t* __restrict__ afea,
            const bf16_t* __restrict__ Wv1T, const bf16_t* __restrict__ Wv2T,
            const bf16_t* __restrict__ Wa1T, const bf16_t* __restrict__ Wa2T,
            bf16_t* __restrict__ v_b1T, bf16_t* __restrict__ v_b2,
            bf16_t* __restrict__ a_b1,  bf16_t* __restrict__ a_b2T) {
  int which = blockIdx.z;
  const bf16_t* X = (which < 2) ? vfea : afea;
  const bf16_t* W = (which == 0) ? Wv1T : (which == 1) ? Wv2T
                   : (which == 2) ? Wa1T : Wa2T;
  int n0 = blockIdx.x * 32, m0 = blockIdx.y * 32;
  Acc4 t = gemm_32x32(X + (size_t)m0 * D_, W + (size_t)n0 * D_, D_, D_);
  int nl, hl; tile_coords(nl, hl);
  const v8f* accs[4] = { &t.c00, &t.c01, &t.c10, &t.c11 };
#pragma unroll
  for (int tm = 0; tm < 2; ++tm)
#pragma unroll
    for (int tn = 0; tn < 2; ++tn) {
      const v8f& c = *accs[tm * 2 + tn];
      int n = n0 + tn * 16 + nl;
#pragma unroll
      for (int r = 0; r < 8; ++r) {
        int m = m0 + tm * 16 + r + 8 * hl;         // global row in [0, B*T)
        float v = fmaxf(c[r], 0.f);
        bf16_t bv = (bf16_t)v;
        int bb = m / T_, tt = m % T_;
        if (which == 0)      v_b1T[((size_t)bb * D_ + n) * T_ + tt] = bv;
        else if (which == 1) v_b2[(size_t)m * D_ + n] = bv;
        else if (which == 2) a_b1[(size_t)m * D_ + n] = bv;
        else                 a_b2T[((size_t)bb * D_ + n) * T_ + tt] = bv;
      }
    }
}

// ---- 3) beta[b][v][a] = relu(scale * v_b2[v,:] . a_b1[a,:]) as bf16.
__global__ __launch_bounds__(32)
void k_beta(const bf16_t* __restrict__ v_b2, const bf16_t* __restrict__ a_b1,
            bf16_t* __restrict__ beta) {
  int b = blockIdx.z;
  int a0 = blockIdx.x * 32, v0 = blockIdx.y * 32;
  const bf16_t* A  = v_b2 + (size_t)b * T_ * D_ + (size_t)v0 * D_;
  const bf16_t* Bm = a_b1 + (size_t)b * T_ * D_ + (size_t)a0 * D_;
  Acc4 t = gemm_32x32(A, Bm, D_, D_);
  const float scale = 0.0625f;               // 1/sqrt(256)
  int nl, hl; tile_coords(nl, hl);
  bf16_t* out = beta + (size_t)b * T_ * T_;
  const v8f* accs[4] = { &t.c00, &t.c01, &t.c10, &t.c11 };
#pragma unroll
  for (int tm = 0; tm < 2; ++tm)
#pragma unroll
    for (int tn = 0; tn < 2; ++tn) {
      const v8f& c = *accs[tm * 2 + tn];
      int a = a0 + tn * 16 + nl;
#pragma unroll
      for (int r = 0; r < 8; ++r) {
        int v = v0 + tm * 16 + r + 8 * hl;
        out[(size_t)v * T_ + a] = (bf16_t)fmaxf(c[r] * scale, 0.f);
      }
    }
}

// ---- 4a) rowsum over audio axis (one workgroup per row, coalesced).
__global__ void k_rowsum(const bf16_t* __restrict__ beta, float* __restrict__ rs) {
  int row = blockIdx.x;                      // [0, B*T)
  const bf16_t* p = beta + (size_t)row * T_;
  float s = 0.f;
  for (int a = threadIdx.x; a < T_; a += 256) s += (float)p[a];
  __shared__ float red[256];
  red[threadIdx.x] = s; __syncthreads();
  for (int st = 128; st > 0; st >>= 1) {
    if (threadIdx.x < st) red[threadIdx.x] += red[threadIdx.x + st];
    __syncthreads();
  }
  if (threadIdx.x == 0) rs[row] = red[0];
}

// ---- 4b) colsum over video axis (thread-per-column, rows coalesced).
__global__ void k_colsum(const bf16_t* __restrict__ beta, float* __restrict__ cs) {
  int b = blockIdx.y;
  int a = blockIdx.x * 256 + threadIdx.x;
  const bf16_t* p = beta + (size_t)b * T_ * T_ + a;
  float s = 0.f;
  for (int v = 0; v < T_; ++v) s += (float)p[(size_t)v * T_];
  cs[b * T_ + a] = s;
}

// ---- 5) gamma: in-place beta -> gamma_va; LDS 64x64 tile transpose -> gamma_av.
__global__ void k_gamma(bf16_t* __restrict__ beta, bf16_t* __restrict__ gamma_av,
                        const float* __restrict__ rs, const float* __restrict__ cs,
                        const float* __restrict__ thr_p) {
  int b = blockIdx.z;
  int a0 = blockIdx.x * 64, v0 = blockIdx.y * 64;
  float thr = *thr_p;
  __shared__ bf16_t tile[64][68];
  bf16_t* bb = beta + (size_t)b * T_ * T_;
#pragma unroll
  for (int i = 0; i < 16; ++i) {
    int idx = threadIdx.x + 256 * i;
    int r = idx >> 6, c = idx & 63;
    float x = (float)bb[(size_t)(v0 + r) * T_ + a0 + c];
    tile[r][c] = (bf16_t)x;
    float g = x / (rs[b * T_ + v0 + r] + EPS_NORM);
    g = (g > thr) ? g : 0.f;
    bb[(size_t)(v0 + r) * T_ + a0 + c] = (bf16_t)g;   // gamma_va in place
  }
  __syncthreads();
  bf16_t* gav = gamma_av + (size_t)b * T_ * T_;
#pragma unroll
  for (int i = 0; i < 16; ++i) {
    int idx = threadIdx.x + 256 * i;
    int ar = idx >> 6, vc = idx & 63;
    float x = (float)tile[vc][ar];
    float g = x / (cs[b * T_ + a0 + ar] + EPS_NORM);
    g = (g > thr) ? g : 0.f;
    gav[(size_t)(a0 + ar) * T_ + v0 + vc] = (bf16_t)g; // coalesced transposed write
  }
}

// ---- 6) pos = G[T,T] @ V (V given transposed as Vt[D][T]); f32 output.
__global__ __launch_bounds__(32)
void k_posgemm(const bf16_t* __restrict__ G, const bf16_t* __restrict__ Vt,
               float* __restrict__ out) {
  int b = blockIdx.z;
  int n0 = blockIdx.x * 32, m0 = blockIdx.y * 32;
  const bf16_t* Gb = G + (size_t)b * T_ * T_ + (size_t)m0 * T_;
  const bf16_t* Vb = Vt + (size_t)b * D_ * T_ + (size_t)n0 * T_;
  Acc4 t = gemm_32x32(Gb, Vb, T_, T_);
  int nl, hl; tile_coords(nl, hl);
  float* ob = out + (size_t)b * T_ * D_;
  const v8f* accs[4] = { &t.c00, &t.c01, &t.c10, &t.c11 };
#pragma unroll
  for (int tm = 0; tm < 2; ++tm)
#pragma unroll
    for (int tn = 0; tn < 2; ++tn) {
      const v8f& c = *accs[tm * 2 + tn];
      int n = n0 + tn * 16 + nl;
#pragma unroll
      for (int r = 0; r < 8; ++r)
        ob[(size_t)(m0 + tm * 16 + r + 8 * hl) * D_ + n] = c[r];
    }
}

// ---- 7) fused dual LayerNorm + average. One workgroup per row, D=256 threads.
__global__ void k_ln(const float* __restrict__ a_fea, const float* __restrict__ v_fea,
                     const float* __restrict__ a_pos, const float* __restrict__ v_pos,
                     const float* __restrict__ lng, const float* __restrict__ lnb,
                     float* __restrict__ out) {
  int row = blockIdx.x;                      // b*T + t
  int d = threadIdx.x;
  size_t o = (size_t)row * D_ + d;
  float xv = v_fea[o] + a_pos[o];
  float xa = a_fea[o] + v_pos[o];
  __shared__ float red[256];
  auto blocksum = [&](float val) -> float {
    red[threadIdx.x] = val; __syncthreads();
    for (int st = 128; st > 0; st >>= 1) {
      if (threadIdx.x < st) red[threadIdx.x] += red[threadIdx.x + st];
      __syncthreads();
    }
    float r = red[0]; __syncthreads();
    return r;
  };
  float sv  = blocksum(xv);
  float svv = blocksum(xv * xv);
  float sa  = blocksum(xa);
  float saa = blocksum(xa * xa);
  const float inv = 1.0f / (float)D_;
  float muv = sv * inv,  varv = svv * inv - muv * muv;
  float mua = sa * inv,  vara = saa * inv - mua * mua;
  float yv = (xv - muv) * rsqrtf(varv + EPS_LN) * lng[d] + lnb[d];
  float ya = (xa - mua) * rsqrtf(vara + EPS_LN) * lng[d] + lnb[d];
  out[o] = 0.5f * (yv + ya);
}

extern "C" void kernel_launch(void* const* d_in, const int* in_sizes, int n_in,
                              void* d_out, int out_size, void* d_ws, size_t ws_size,
                              hipStream_t stream) {
  (void)in_sizes; (void)n_in; (void)out_size; (void)ws_size;
  const float* a_fea = (const float*)d_in[0];
  const float* v_fea = (const float*)d_in[1];
  const float* Wv1   = (const float*)d_in[2];
  const float* Wv2   = (const float*)d_in[3];
  const float* Wa1   = (const float*)d_in[4];
  const float* Wa2   = (const float*)d_in[5];
  const float* lng   = (const float*)d_in[6];
  const float* lnb   = (const float*)d_in[7];
  const float* thr   = (const float*)d_in[8];
  float* out = (float*)d_out;

  // ---- workspace carve-up (256B aligned) ----
  char* base = (char*)d_ws;
  size_t off = 0;
  auto alloc = [&](size_t bytes) -> char* {
    char* p = base + off;
    off += (bytes + 255) & ~(size_t)255;
    return p;
  };
  const size_t NA = (size_t)B_ * T_ * D_;       // 4,194,304 activations
  const size_t NB = (size_t)B_ * T_ * T_;       // 33,554,432 beta elements
  bf16_t* afea_bf = (bf16_t*)alloc(NA * 2);
  bf16_t* vfea_bf = (bf16_t*)alloc(NA * 2);
  bf16_t* Wv1T = (bf16_t*)alloc((size_t)D_ * D_ * 2);
  bf16_t* Wv2T = (bf16_t*)alloc((size_t)D_ * D_ * 2);
  bf16_t* Wa1T = (bf16_t*)alloc((size_t)D_ * D_ * 2);
  bf16_t* Wa2T = (bf16_t*)alloc((size_t)D_ * D_ * 2);
  bf16_t* v_b1T = (bf16_t*)alloc(NA * 2);       // [b][h][t]
  bf16_t* v_b2  = (bf16_t*)alloc(NA * 2);       // [b][t][h]
  bf16_t* a_b1  = (bf16_t*)alloc(NA * 2);       // [b][t][h]
  bf16_t* a_b2T = (bf16_t*)alloc(NA * 2);       // [b][h][t]
  bf16_t* beta  = (bf16_t*)alloc(NB * 2);       // becomes gamma_va in place
  bf16_t* gamma_av = (bf16_t*)alloc(NB * 2);
  float* rs    = (float*)alloc((size_t)B_ * T_ * 4);
  float* cs    = (float*)alloc((size_t)B_ * T_ * 4);
  float* a_pos = (float*)alloc(NA * 4);
  float* v_pos = (float*)alloc(NA * 4);

  // 1) convert to bf16 (+ weight transposes)
  k_convert<<<(int)((NA + 255) / 256), 256, 0, stream>>>(
      a_fea, v_fea, Wv1, Wv2, Wa1, Wa2,
      afea_bf, vfea_bf, Wv1T, Wv2T, Wa1T, Wa2T);

  // 2) four projection GEMMs (32x32 tiles per wave)
  k_proj<<<dim3(D_ / 32, (B_ * T_) / 32, 4), 32, 0, stream>>>(
      vfea_bf, afea_bf, Wv1T, Wv2T, Wa1T, Wa2T, v_b1T, v_b2, a_b1, a_b2T);

  // 3) beta GEMM (relu + scale)
  k_beta<<<dim3(T_ / 32, T_ / 32, B_), 32, 0, stream>>>(v_b2, a_b1, beta);

  // 4) row / col sums
  k_rowsum<<<B_ * T_, 256, 0, stream>>>(beta, rs);
  k_colsum<<<dim3(T_ / 256, B_), 256, 0, stream>>>(beta, cs);

  // 5) gammas (in-place + LDS transpose)
  k_gamma<<<dim3(T_ / 64, T_ / 64, B_), 256, 0, stream>>>(beta, gamma_av, rs, cs, thr);

  // 6) attention-apply GEMMs (32x32 tiles per wave)
  k_posgemm<<<dim3(D_ / 32, T_ / 32, B_), 32, 0, stream>>>(beta, a_b2T, a_pos);
  k_posgemm<<<dim3(D_ / 32, T_ / 32, B_), 32, 0, stream>>>(gamma_av, v_b1T, v_pos);

  // 7) fused dual LayerNorm + average
  k_ln<<<B_ * T_, 256, 0, stream>>>(a_fea, v_fea, a_pos, v_pos, lng, lnb, out);
}
